// LowRank_2473901162708
// MI455X (gfx1250) — matrix operand order repaired and checked by reading, not compiled, not run on hardware
//
#include <hip/hip_runtime.h>
#include <cstdint>
#include <cstddef>

typedef __attribute__((ext_vector_type(16))) _Float16 v16h;
typedef __attribute__((ext_vector_type(8)))  _Float16 v8h;
typedef __attribute__((ext_vector_type(4)))  _Float16 v4h;
typedef __attribute__((ext_vector_type(8)))  float    v8f;

#define B_   64
#define M_   1024
#define E_   1024
#define H_   8
#define D_   128
#define MID_ 64
#define ALPHA_ 1.3f
#define EPS_   1e-5f

__device__ __forceinline__ float celu_f(float x) {
    return x > 0.f ? x : ALPHA_ * (expf(x * (1.f / ALPHA_)) - 1.f);
}

// ---------------------------------------------------------------------------
// W (E x E, f32, k-major) -> WT (E x E, f16, n-major) so GEMM B-fragments are
// contiguous-K loads.
// ---------------------------------------------------------------------------
__global__ void k_transpose_w(const float* __restrict__ W, _Float16* __restrict__ WT) {
    __shared__ float tile[32][33];
    const int n0 = blockIdx.x * 32, k0 = blockIdx.y * 32;
    const int t = threadIdx.x;
    const int r = t >> 5, c = t & 31;
#pragma unroll
    for (int i = 0; i < 4; i++)
        tile[r + 8 * i][c] = W[(size_t)(k0 + r + 8 * i) * E_ + n0 + c];
    __syncthreads();
#pragma unroll
    for (int i = 0; i < 4; i++)
        WT[(size_t)(n0 + r + 8 * i) * E_ + k0 + c] = (_Float16)tile[c][r + 8 * i];
}

// ---------------------------------------------------------------------------
// Small projection (64 rows): out = groupnorm(celu(x @ W + b)) in f32.
// One block per row, 256 threads, 4 output columns each.
// ---------------------------------------------------------------------------
__global__ __launch_bounds__(256) void k_small_proj(
    const float* __restrict__ x, const float* __restrict__ W,
    const float* __restrict__ bias, const float* __restrict__ gamma,
    const float* __restrict__ beta, float* __restrict__ out)
{
    __shared__ float xs[E_];
    __shared__ float ys[E_];
    __shared__ float gs[H_], gq[H_];
    const int b = blockIdx.x, t = threadIdx.x;
#pragma unroll
    for (int i = 0; i < 4; i++) xs[t + 256 * i] = x[(size_t)b * E_ + t + 256 * i];
    if (t < H_) { gs[t] = 0.f; gq[t] = 0.f; }
    __syncthreads();

    float acc[4] = {0.f, 0.f, 0.f, 0.f};
    for (int k = 0; k < E_; k++) {
        const float xv = xs[k];
        const float* wr = W + (size_t)k * E_;
#pragma unroll
        for (int i = 0; i < 4; i++) acc[i] = fmaf(xv, wr[t + 256 * i], acc[i]);
    }
#pragma unroll
    for (int i = 0; i < 4; i++) {
        const int c = t + 256 * i;
        const float v = celu_f(acc[i] + bias[c]);
        ys[c] = v;
        atomicAdd(&gs[c >> 7], v);
        atomicAdd(&gq[c >> 7], v * v);
    }
    __syncthreads();
    const float invD = 1.f / (float)D_;
#pragma unroll
    for (int i = 0; i < 4; i++) {
        const int c = t + 256 * i, g = c >> 7;
        const float mu  = gs[g] * invD;
        const float var = gq[g] * invD - mu * mu;
        out[(size_t)b * E_ + c] = (ys[c] - mu) * rsqrtf(var + EPS_) * gamma[c] + beta[c];
    }
}

// wsum[j] = sum_d W_basic[d][j]
__global__ void k_wsum(const float* __restrict__ Wb, float* __restrict__ wsum) {
    const int j = threadIdx.x;
    float s = 0.f;
    for (int d = 0; d < D_; d++) s += Wb[(size_t)d * MID_ + j];
    wsum[j] = s;
}

// ---------------------------------------------------------------------------
// Big projection GEMM: out_f16 = groupnorm(celu(A @ W + b))
//   A : (65536 x 1024) f32 (converted to f16 on the fly)
//   WT: (1024 x 1024) f16, n-major
// Tile: BM=128 x BN=256 x BK=32; 8 waves, each 64x64 via 16 WMMA / K-step.
// ---------------------------------------------------------------------------
__global__ __launch_bounds__(256) void k_gemm_proj(
    const float* __restrict__ A, const _Float16* __restrict__ WT,
    const float* __restrict__ bias, const float* __restrict__ gamma,
    const float* __restrict__ beta, _Float16* __restrict__ out)
{
    // Row stride 40 halfs = 80 B: 16B-aligned for b128 and conflict-free
    // (bank step 20, gcd(20,64)=4 -> 16-lane period covers distinct banks).
    __shared__ _Float16 Alds[2][128 * 40];   // 20480 B
    __shared__ _Float16 Wlds[2][256 * 40];   // 40960 B
    __shared__ float rowsum[128][2];
    __shared__ float rowsq [128][2];

    const int t    = threadIdx.x;
    const int lane = t & 31, wv = t >> 5;
    const int l16  = lane & 15, lh = lane >> 4;
    const int wm   = (wv >> 2) * 64;          // 0 or 64
    const int wn   = (wv & 3) * 64;           // 0,64,128,192
    const int m0   = blockIdx.y * 128;
    const int n0   = blockIdx.x * 256;

    if (t < 128) {
        rowsum[t][0] = 0.f; rowsum[t][1] = 0.f;
        rowsq [t][0] = 0.f; rowsq [t][1] = 0.f;
    }

    float bnv[4], gnv[4], bev[4];
#pragma unroll
    for (int nt = 0; nt < 4; nt++) {
        const int n = n0 + wn + nt * 16 + l16;
        bnv[nt] = bias[n]; gnv[nt] = gamma[n]; bev[nt] = beta[n];
    }

    v8f acc[4][4];
    const v8f vzero = {0.f, 0.f, 0.f, 0.f, 0.f, 0.f, 0.f, 0.f};
#pragma unroll
    for (int i = 0; i < 4; i++)
#pragma unroll
        for (int j = 0; j < 4; j++) acc[i][j] = vzero;

    float4 areg[4];
    v8h    wreg[4];

    auto loadA = [&](int k0) {
#pragma unroll
        for (int i = 0; i < 4; i++) {
            const int c = t + 256 * i;
            const int row = c >> 3, col = c & 7;
            areg[i] = *reinterpret_cast<const float4*>(
                A + (size_t)(m0 + row) * E_ + k0 + col * 4);
        }
    };
    auto loadW = [&](int k0) {
        const _Float16* base = WT + (size_t)(n0 + t) * E_ + k0;
#pragma unroll
        for (int i = 0; i < 4; i++)
            wreg[i] = *reinterpret_cast<const v8h*>(base + i * 8);
    };
    auto stageA = [&](int buf) {
#pragma unroll
        for (int i = 0; i < 4; i++) {
            const int c = t + 256 * i;
            const int row = c >> 3, col = c & 7;
            v4h hv = { (_Float16)areg[i].x, (_Float16)areg[i].y,
                       (_Float16)areg[i].z, (_Float16)areg[i].w };
            *reinterpret_cast<v4h*>(&Alds[buf][row * 40 + col * 4]) = hv;
        }
    };
    auto stageW = [&](int buf) {
#pragma unroll
        for (int i = 0; i < 4; i++)
            *reinterpret_cast<v8h*>(&Wlds[buf][t * 40 + i * 8]) = wreg[i];
    };
    auto compute = [&](int buf) {
        v16h af[4];
#pragma unroll
        for (int mt = 0; mt < 4; mt++) {
            // A frag: elems 0-7 -> K = lh*8 + e ; elems 8-15 -> K = 16 + lh*8 + e
            const _Float16* p = &Alds[buf][(wm + mt * 16 + l16) * 40 + lh * 8];
            v16h a;
            reinterpret_cast<v8h*>(&a)[0] = *reinterpret_cast<const v8h*>(p);
            reinterpret_cast<v8h*>(&a)[1] = *reinterpret_cast<const v8h*>(p + 16);
            af[mt] = a;
        }
#pragma unroll
        for (int nt = 0; nt < 4; nt++) {
            // B frag: N = l16, elems 0-7 -> K = lh*8 + e ; elems 8-15 -> K = 16 + lh*8 + e
            const _Float16* q = &Wlds[buf][(wn + nt * 16 + l16) * 40 + lh * 8];
            v16h bf;
            reinterpret_cast<v8h*>(&bf)[0] = *reinterpret_cast<const v8h*>(q);
            reinterpret_cast<v8h*>(&bf)[1] = *reinterpret_cast<const v8h*>(q + 16);
#pragma unroll
            for (int mt = 0; mt < 4; mt++)
                acc[mt][nt] = __builtin_amdgcn_wmma_f32_16x16x32_f16(
                    false, af[mt], false, bf, (short)0, acc[mt][nt], false, false);
        }
    };

    loadA(0); loadW(0);
    stageA(0); stageW(0);
    __syncthreads();

#pragma unroll 1
    for (int s = 0; s < 32; s++) {
        if (s < 31) { loadA((s + 1) * 32); loadW((s + 1) * 32); }   // prefetch overlaps WMMA
        compute(s & 1);
        if (s < 31) {
            __syncthreads();
            stageA((s + 1) & 1); stageW((s + 1) & 1);
            __syncthreads();
        }
    }

    // Epilogue: bias + CELU, groupnorm stats via LDS float atomics.
    const int g = wn >> 7;   // 128-wide group index within the 256-wide tile
#pragma unroll
    for (int mt = 0; mt < 4; mt++) {
#pragma unroll
        for (int v = 0; v < 8; v++) {
            float s = 0.f, s2 = 0.f;
#pragma unroll
            for (int nt = 0; nt < 4; nt++) {
                float xv = acc[mt][nt][v] + bnv[nt];
                xv = celu_f(xv);
                acc[mt][nt][v] = xv;
                s += xv; s2 += xv * xv;
            }
            const int ml = wm + mt * 16 + v + 8 * lh;
            atomicAdd(&rowsum[ml][g], s);
            atomicAdd(&rowsq [ml][g], s2);
        }
    }
    __syncthreads();

    const float invD = 1.f / (float)D_;
#pragma unroll
    for (int mt = 0; mt < 4; mt++) {
#pragma unroll
        for (int v = 0; v < 8; v++) {
            const int ml = wm + mt * 16 + v + 8 * lh;
            const float mu  = rowsum[ml][g] * invD;
            const float var = rowsq [ml][g] * invD - mu * mu;
            const float rs  = rsqrtf(var + EPS_);
            const size_t rowg = (size_t)(m0 + ml) * E_;
#pragma unroll
            for (int nt = 0; nt < 4; nt++) {
                const int n = n0 + wn + nt * 16 + l16;
                out[rowg + n] = (_Float16)((acc[mt][nt][v] - mu) * rs * gnv[nt] + bev[nt]);
            }
        }
    }
}

// ---------------------------------------------------------------------------
// Fused attention tail: one block per (b,h).
// scores -> rank-1 hidden -> pool / alpha_sp -> softmax -> alpha_ch -> v2_att.
// ---------------------------------------------------------------------------
__global__ __launch_bounds__(256) void k_attn(
    const _Float16* __restrict__ kproj, const _Float16* __restrict__ v2proj,
    const float* __restrict__ qproj, const float* __restrict__ v1proj,
    const float* __restrict__ mask,  const float* __restrict__ wsum,
    const float* __restrict__ b_basic, const float* __restrict__ W_sp,
    const float* __restrict__ b_sp, const float* __restrict__ W_ch,
    const float* __restrict__ b_ch, float* __restrict__ out)
{
    __shared__ float qv[D_];
    __shared__ float lg[M_];
    __shared__ float pool[MID_];
    __shared__ float wsm[MID_], bbs[MID_], wsp[MID_];
    __shared__ float red[16];
    __shared__ float vpart[D_];

    const int h = blockIdx.x, b = blockIdx.y;
    const int t = threadIdx.x;
    const float scaling = 0.08838834764831845f;   // 128^-0.5

    if (t < MID_) { pool[t] = 0.f; wsm[t] = wsum[t]; bbs[t] = b_basic[t]; wsp[t] = W_sp[t]; }
    if (t < D_)   qv[t] = qproj[(size_t)b * E_ + h * D_ + t];
    if (t < 16)   red[t] = 0.f;
    __syncthreads();

    float lp[MID_];
#pragma unroll
    for (int j = 0; j < MID_; j++) lp[j] = 0.f;
    float msuml = 0.f;
    const float bsp0 = b_sp[0];

    for (int i = 0; i < 4; i++) {
        const int m = t + 256 * i;
        const _Float16* kr = kproj + (size_t)(b * M_ + m) * E_ + h * D_;
        float s = 0.f;
#pragma unroll
        for (int c = 0; c < 16; c++) {
            v8h kk = *reinterpret_cast<const v8h*>(kr + c * 8);
#pragma unroll
            for (int j = 0; j < 8; j++) s = fmaf(qv[c * 8 + j], (float)kk[j], s);
        }
        s *= scaling;
        const float mv = mask[(size_t)b * M_ + m];
        msuml += mv;
        float sp = 0.f;
#pragma unroll 8
        for (int j = 0; j < MID_; j++) {
            const float hd = fmaxf(fmaf(s, wsm[j], bbs[j]), 0.f);
            sp    = fmaf(hd, wsp[j], sp);
            lp[j] = fmaf(hd, mv, lp[j]);
        }
        lg[m] = (mv == 0.f) ? -1.0e9f : (sp + bsp0);
    }
#pragma unroll
    for (int j = 0; j < MID_; j++) atomicAdd(&pool[j], lp[j]);
    atomicAdd(&red[0], msuml);
    __syncthreads();

    // softmax: max
    float mx = -3.0e38f;
    for (int i = 0; i < 4; i++) mx = fmaxf(mx, lg[t + 256 * i]);
#pragma unroll
    for (int off = 16; off > 0; off >>= 1) mx = fmaxf(mx, __shfl_xor(mx, off, 32));
    if ((t & 31) == 0) red[1 + (t >> 5)] = mx;
    __syncthreads();
    if (t == 0) {
        float m2 = red[1];
        for (int w = 1; w < 8; w++) m2 = fmaxf(m2, red[1 + w]);
        red[9] = m2;
    }
    __syncthreads();
    mx = red[9];

    // softmax: exp + sum
    float esl = 0.f;
    for (int i = 0; i < 4; i++) {
        const int m = t + 256 * i;
        const float e = expf(lg[m] - mx);
        lg[m] = e;
        esl += e;
    }
#pragma unroll
    for (int off = 16; off > 0; off >>= 1) esl += __shfl_xor(esl, off, 32);
    if ((t & 31) == 0) atomicAdd(&red[10], esl);
    __syncthreads();

    const float msum    = red[0];
    const float esum    = red[10];
    const float invmsum = 1.f / msum;

    float ach = 0.f, v1v = 0.f;
    if (t < D_) {
        float pa = b_ch[t];
#pragma unroll 8
        for (int j = 0; j < MID_; j++)
            pa = fmaf(pool[j] * invmsum, W_ch[(size_t)j * D_ + t], pa);
        ach = 1.f / (1.f + expf(-pa));
        v1v = v1proj[(size_t)b * E_ + h * D_ + t];
    }

    // v2_att: threads 0-127 handle even m, 128-255 odd m; d = t & 127.
    const int d  = t & (D_ - 1);
    const int hf = t >> 7;
    float va = 0.f;
    for (int m = hf; m < M_; m += 2)
        va = fmaf(lg[m], (float)v2proj[(size_t)(b * M_ + m) * E_ + h * D_ + d], va);
    if (hf == 0) vpart[d] = va;
    __syncthreads();
    if (hf == 1) vpart[d] += va;
    __syncthreads();

    if (t < D_)
        out[(size_t)b * E_ + h * D_ + t] = v1v * (vpart[t] / esum) * ach;
}

// ---------------------------------------------------------------------------
extern "C" void kernel_launch(void* const* d_in, const int* in_sizes, int n_in,
                              void* d_out, int out_size, void* d_ws, size_t ws_size,
                              hipStream_t stream)
{
    (void)in_sizes; (void)n_in; (void)out_size; (void)ws_size;

    const float* query     = (const float*)d_in[0];
    const float* key_feats = (const float*)d_in[1];
    const float* mask      = (const float*)d_in[2];
    const float* value1    = (const float*)d_in[3];
    const float* value2    = (const float*)d_in[4];
    const float* W_q  = (const float*)d_in[5];
    const float* b_q  = (const float*)d_in[6];
    const float* g_q  = (const float*)d_in[7];
    const float* be_q = (const float*)d_in[8];
    const float* W_k  = (const float*)d_in[9];
    const float* b_k  = (const float*)d_in[10];
    const float* g_k  = (const float*)d_in[11];
    const float* be_k = (const float*)d_in[12];
    const float* W_v1  = (const float*)d_in[13];
    const float* b_v1  = (const float*)d_in[14];
    const float* g_v1  = (const float*)d_in[15];
    const float* be_v1 = (const float*)d_in[16];
    const float* W_v2  = (const float*)d_in[17];
    const float* b_v2  = (const float*)d_in[18];
    const float* g_v2  = (const float*)d_in[19];
    const float* be_v2 = (const float*)d_in[20];
    const float* W_basic = (const float*)d_in[21];
    const float* b_basic = (const float*)d_in[22];
    const float* W_sp    = (const float*)d_in[23];
    const float* b_sp    = (const float*)d_in[24];
    const float* W_ch    = (const float*)d_in[25];
    const float* b_ch    = (const float*)d_in[26];

    char* w = (char*)d_ws;
    size_t off = 0;
    _Float16* kproj  = (_Float16*)(w + off); off += (size_t)B_ * M_ * E_ * 2;  // 128 MB
    _Float16* v2proj = (_Float16*)(w + off); off += (size_t)B_ * M_ * E_ * 2;  // 128 MB
    _Float16* WTk    = (_Float16*)(w + off); off += (size_t)E_ * E_ * 2;       // 2 MB
    _Float16* WTv2   = (_Float16*)(w + off); off += (size_t)E_ * E_ * 2;       // 2 MB
    float* qproj  = (float*)(w + off); off += (size_t)B_ * E_ * 4;             // 256 KB
    float* v1proj = (float*)(w + off); off += (size_t)B_ * E_ * 4;             // 256 KB
    float* wsum   = (float*)(w + off); off += 256;

    k_transpose_w<<<dim3(32, 32), 256, 0, stream>>>(W_k,  WTk);
    k_transpose_w<<<dim3(32, 32), 256, 0, stream>>>(W_v2, WTv2);
    k_small_proj<<<B_, 256, 0, stream>>>(query,  W_q,  b_q,  g_q,  be_q,  qproj);
    k_small_proj<<<B_, 256, 0, stream>>>(value1, W_v1, b_v1, g_v1, be_v1, v1proj);
    k_wsum<<<1, 64, 0, stream>>>(W_basic, wsum);
    k_gemm_proj<<<dim3(4, 512), 256, 0, stream>>>(key_feats, WTk,  b_k,  g_k,  be_k,  kproj);
    k_gemm_proj<<<dim3(4, 512), 256, 0, stream>>>(value2,    WTv2, b_v2, g_v2, be_v2, v2proj);
    k_attn<<<dim3(H_, B_), 256, 0, stream>>>(kproj, v2proj, qproj, v1proj, mask, wsum,
                                             b_basic, W_sp, b_sp, W_ch, b_ch,
                                             (float*)d_out);
}
